// RGCN_31250182046566
// MI455X (gfx1250) — matrix operand order repaired and checked by reading, not compiled, not run on hardware
//
#include <hip/hip_runtime.h>
#include <math.h>

typedef _Float16 half_t;
typedef __attribute__((ext_vector_type(16))) _Float16 v16h;
typedef __attribute__((ext_vector_type(8)))  _Float16 h8;
typedef __attribute__((ext_vector_type(8)))  float    v8f;
typedef __attribute__((ext_vector_type(4)))  float    f4;
typedef __attribute__((ext_vector_type(4)))  unsigned u4;

#define NNODES 50000
#define NEDGES 600000

// ---------------- helpers ----------------
__device__ inline float wsum(float v) {
#pragma unroll
  for (int m = 16; m >= 1; m >>= 1) v += __shfl_xor(v, m, 32);
  return v;
}
__device__ inline float wmax(float v) {
#pragma unroll
  for (int m = 16; m >= 1; m >>= 1) v = fmaxf(v, __shfl_xor(v, m, 32));
  return v;
}
__device__ inline void atomAddF(float* p, float v) {
  __hip_atomic_fetch_add(p, v, __ATOMIC_RELAXED, __HIP_MEMORY_SCOPE_AGENT);
}
__device__ inline void atomMaxU(unsigned* p, unsigned v) {
  __hip_atomic_fetch_max(p, v, __ATOMIC_RELAXED, __HIP_MEMORY_SCOPE_AGENT);
}
// order-preserving float<->u32 key (no NaNs in data path)
__device__ inline unsigned encf(float f) {
  unsigned u = __float_as_uint(f);
  return (u & 0x80000000u) ? ~u : (u | 0x80000000u);
}
__device__ inline float decf(unsigned k) {
  unsigned u = (k & 0x80000000u) ? (k & 0x7FFFFFFFu) : ~k;
  return __uint_as_float(u);
}
#define ENC_NEG_INF 0x007FFFFFu  // encf(-inf)

// ---------------- prep kernels ----------------
__global__ void f32_to_f16_k(const float* __restrict__ s, half_t* __restrict__ d, size_t n) {
  size_t i = (size_t)blockIdx.x * blockDim.x + threadIdx.x;
  if (i < n) d[i] = (half_t)s[i];
}
// dst[n][k] = (n<N) ? src[k][n] : 0  (grid.x = Npad, block.x = K)
__global__ void pack_wT_k(const float* __restrict__ src, half_t* __restrict__ dst, int K, int N) {
  int n = blockIdx.x, k = threadIdx.x;
  float v = (n < N) ? src[(size_t)k * N + n] : 0.0f;
  dst[(size_t)n * K + k] = (half_t)v;
}
__global__ void fill_u32_k(unsigned* __restrict__ p, unsigned v, size_t n) {
  size_t i = (size_t)blockIdx.x * blockDim.x + threadIdx.x;
  if (i < n) p[i] = v;
}
// out[node][c] = H[node][c] + bias[c]   (H has 512 cols, root in cols 0..127)
__global__ void conv_init_k(const float* __restrict__ H, const float* __restrict__ bias,
                            float* __restrict__ out, int M) {
  size_t i = (size_t)blockIdx.x * blockDim.x + threadIdx.x;
  if (i >= (size_t)M * 128) return;
  int c = (int)(i & 127);
  size_t node = i >> 7;
  out[i] = H[node * 512 + c] + bias[c];
}

// ---------------- WMMA GEMM: C[M][N] = A[M][K](f16) x Bt[N][K](f16) ----------------
// one wave computes a 16 x (16*NT) tile; layouts per CDNA5 ISA 7.12.2
template <int NT>
__global__ void gemm_wmma_k(const half_t* __restrict__ A, const half_t* __restrict__ Bt,
                            float* __restrict__ C, int M, int N, int K) {
  int wave = blockIdx.x * (blockDim.x >> 5) + (threadIdx.x >> 5);
  int lane = threadIdx.x & 31;
  int nTiles = N / (16 * NT);
  int total = (M / 16) * nTiles;
  if (wave >= total) return;
  int mb = wave / nTiles;
  int nb = wave % nTiles;
  int lrow = lane & 15;
  int halfId = lane >> 4;

  v8f acc[NT];
#pragma unroll
  for (int t = 0; t < NT; t++) acc[t] = (v8f)(0.0f);

  const half_t* arow = A + (size_t)(mb * 16 + lrow) * K;
  for (int kc = 0; kc < K; kc += 32) {
    h8 alo = *(const h8*)(arow + kc + halfId * 8);
    h8 ahi = *(const h8*)(arow + kc + 16 + halfId * 8);
    v16h a;
#pragma unroll
    for (int i = 0; i < 8; i++) { a[i] = alo[i]; a[8 + i] = ahi[i]; }
#pragma unroll
    for (int t = 0; t < NT; t++) {
      int col = nb * (16 * NT) + t * 16 + lrow;
      const half_t* bcol = Bt + (size_t)col * K + kc + halfId * 8;
      h8 blo = *(const h8*)(bcol);
      h8 bhi = *(const h8*)(bcol + 16);
      v16h b;
#pragma unroll
      for (int i = 0; i < 8; i++) { b[i] = blo[i]; b[8 + i] = bhi[i]; }
      acc[t] = __builtin_amdgcn_wmma_f32_16x16x32_f16(false, a, false, b,
                                                      (short)0, acc[t], false, false);
    }
  }
#pragma unroll
  for (int t = 0; t < NT; t++) {
    int col = nb * (16 * NT) + t * 16 + lrow;
#pragma unroll
    for (int r = 0; r < 8; r++) {
      int m = mb * 16 + r + 8 * halfId;
      C[(size_t)m * N + col] = acc[t][r];
    }
  }
}

// ---------------- edge scatter kernels (wave per edge) ----------------
__global__ void scatter_add_k(const float* __restrict__ H, const int* __restrict__ ei,
                              const int* __restrict__ et, float* __restrict__ out, int E) {
  int e = blockIdx.x * (blockDim.x >> 5) + (threadIdx.x >> 5);
  int lane = threadIdx.x & 31;
  if (e >= E) return;
  int src = ei[e], dst = ei[E + e], r = et[e];
  f4 m = *(const f4*)(H + (size_t)src * 512 + 128 + (size_t)r * 128 + lane * 4);
  float* o = out + (size_t)dst * 128 + lane * 4;
  atomAddF(o + 0, m[0]); atomAddF(o + 1, m[1]);
  atomAddF(o + 2, m[2]); atomAddF(o + 3, m[3]);
}
__global__ void scatter_max_k(const float* __restrict__ H, const int* __restrict__ ei,
                              const int* __restrict__ et, unsigned* __restrict__ agg,
                              int E, int M) {
  int e = blockIdx.x * (blockDim.x >> 5) + (threadIdx.x >> 5);
  int lane = threadIdx.x & 31;
  if (e >= E) return;
  int src = ei[e], dst = ei[E + e], r = et[e];
  f4 m = *(const f4*)(H + (size_t)src * 512 + 128 + (size_t)r * 128 + lane * 4);
  unsigned* o = agg + ((size_t)r * M + dst) * 128 + lane * 4;
  atomMaxU(o + 0, encf(m[0])); atomMaxU(o + 1, encf(m[1]));
  atomMaxU(o + 2, encf(m[2])); atomMaxU(o + 3, encf(m[3]));
}

// ---------------- fused epilogues (wave per node row) ----------------
__global__ void ln_relu_k(float* __restrict__ x, half_t* __restrict__ xh,
                          const float* __restrict__ g, const float* __restrict__ b, int M) {
  int row = blockIdx.x * (blockDim.x >> 5) + (threadIdx.x >> 5);
  int lane = threadIdx.x & 31;
  if (row >= M) return;
  float* p = x + (size_t)row * 128 + lane * 4;
  f4 v = *(const f4*)p;
  float s = v[0] + v[1] + v[2] + v[3];
  float s2 = v[0]*v[0] + v[1]*v[1] + v[2]*v[2] + v[3]*v[3];
  s = wsum(s); s2 = wsum(s2);
  float mu = s * (1.0f / 128.0f);
  float inv = rsqrtf(s2 * (1.0f / 128.0f) - mu * mu + 1e-5f);
  f4 o;
#pragma unroll
  for (int i = 0; i < 4; i++) {
    float t = (v[i] - mu) * inv * g[lane * 4 + i] + b[lane * 4 + i];
    o[i] = fmaxf(t, 0.0f);
  }
  *(f4*)p = o;
  half_t* hp = xh + (size_t)row * 128 + lane * 4;
#pragma unroll
  for (int i = 0; i < 4; i++) hp[i] = (half_t)o[i];
}

__global__ void conv2_fin_k(const float* __restrict__ out2, const unsigned* __restrict__ agg,
                            const float* __restrict__ x1, const float* __restrict__ g,
                            const float* __restrict__ b, half_t* __restrict__ h16, int M) {
  int row = blockIdx.x * (blockDim.x >> 5) + (threadIdx.x >> 5);
  int lane = threadIdx.x & 31;
  if (row >= M) return;
  f4 v = *(const f4*)(out2 + (size_t)row * 128 + lane * 4);
#pragma unroll
  for (int r = 0; r < 3; r++) {
    u4 k = *(const u4*)(agg + ((size_t)r * M + row) * 128 + lane * 4);
#pragma unroll
    for (int i = 0; i < 4; i++) {
      float a = decf(k[i]);
      if (!isfinite(a)) a = 0.0f;   // empty segment -> 0
      v[i] += a;
    }
  }
  float s = v[0] + v[1] + v[2] + v[3];
  float s2 = v[0]*v[0] + v[1]*v[1] + v[2]*v[2] + v[3]*v[3];
  s = wsum(s); s2 = wsum(s2);
  float mu = s * (1.0f / 128.0f);
  float inv = rsqrtf(s2 * (1.0f / 128.0f) - mu * mu + 1e-5f);
  f4 x1v = *(const f4*)(x1 + (size_t)row * 128 + lane * 4);
  half_t* hp = h16 + (size_t)row * 128 + lane * 4;
#pragma unroll
  for (int i = 0; i < 4; i++) {
    float t = (v[i] - mu) * inv * g[lane * 4 + i] + b[lane * 4 + i];
    t = fmaxf(t, 0.0f) + 0.2f * x1v[i];
    hp[i] = (half_t)t;
  }
}

template <int W>
__global__ void ln_gelu_k(const float* __restrict__ in, const float* __restrict__ bias,
                          const float* __restrict__ g, const float* __restrict__ b,
                          half_t* __restrict__ outh, int M) {
  constexpr int EPL = W / 32;
  int row = blockIdx.x * (blockDim.x >> 5) + (threadIdx.x >> 5);
  int lane = threadIdx.x & 31;
  if (row >= M) return;
  const float* p = in + (size_t)row * W + lane * EPL;
  float v[EPL], s = 0.0f, s2 = 0.0f;
#pragma unroll
  for (int i = 0; i < EPL; i++) {
    v[i] = p[i] + bias[lane * EPL + i];
    s += v[i]; s2 += v[i] * v[i];
  }
  s = wsum(s); s2 = wsum(s2);
  float mu = s * (1.0f / (float)W);
  float inv = rsqrtf(s2 * (1.0f / (float)W) - mu * mu + 1e-5f);
  half_t* hp = outh + (size_t)row * W + lane * EPL;
#pragma unroll
  for (int i = 0; i < EPL; i++) {
    float t = (v[i] - mu) * inv * g[lane * EPL + i] + b[lane * EPL + i];
    float ge = 0.5f * t * (1.0f + erff(t * 0.70710678118654752f));
    hp[i] = (half_t)ge;
  }
}

__global__ void lsm_k(const float* __restrict__ C3, const float* __restrict__ bias,
                      float* __restrict__ out, int M) {
  int row = blockIdx.x * (blockDim.x >> 5) + (threadIdx.x >> 5);
  int lane = threadIdx.x & 31;
  if (row >= M) return;
  const float* p = C3 + (size_t)row * 48;
  float v0 = p[lane] + bias[lane];
  float v1 = (lane < 8) ? (p[32 + lane] + bias[32 + lane]) : -INFINITY;
  float m = wmax(fmaxf(v0, v1));
  float s = __expf(v0 - m) + ((lane < 8) ? __expf(v1 - m) : 0.0f);
  s = wsum(s);
  float l = m + __logf(s);
  out[(size_t)row * 40 + lane] = v0 - l;
  if (lane < 8) out[(size_t)row * 40 + 32 + lane] = v1 - l;
}

// ---------------- host ----------------
extern "C" void kernel_launch(void* const* d_in, const int* in_sizes, int n_in,
                              void* d_out, int out_size, void* d_ws, size_t ws_size,
                              hipStream_t stream) {
  const float* x       = (const float*)d_in[0];
  const int*   ei      = (const int*)d_in[1];
  const int*   et      = (const int*)d_in[2];
  const float* w1_rel  = (const float*)d_in[3];
  const float* w1_root = (const float*)d_in[4];
  const float* b1      = (const float*)d_in[5];
  const float* ln1_g   = (const float*)d_in[6];
  const float* ln1_b   = (const float*)d_in[7];
  const float* w2_rel  = (const float*)d_in[8];
  const float* w2_root = (const float*)d_in[9];
  const float* b2      = (const float*)d_in[10];
  const float* ln2_g   = (const float*)d_in[11];
  const float* ln2_b   = (const float*)d_in[12];
  const float* cw1     = (const float*)d_in[13];
  const float* cb1     = (const float*)d_in[14];
  const float* cln1_g  = (const float*)d_in[15];
  const float* cln1_b  = (const float*)d_in[16];
  const float* cw2     = (const float*)d_in[17];
  const float* cb2     = (const float*)d_in[18];
  const float* cln2_g  = (const float*)d_in[19];
  const float* cln2_b  = (const float*)d_in[20];
  const float* cw3     = (const float*)d_in[21];
  const float* cb3     = (const float*)d_in[22];
  float* out = (float*)d_out;

  char* ws = (char*)d_ws;
  size_t off = 0;
  auto take = [&](size_t bytes) { char* p = ws + off; off = (off + bytes + 255) & ~(size_t)255; return p; };
  half_t* XH   = (half_t*)take((size_t)NNODES * 128 * 2);
  half_t* X1H  = (half_t*)take((size_t)NNODES * 128 * 2);
  half_t* H16C = (half_t*)take((size_t)NNODES * 128 * 2);
  half_t* Y16  = (half_t*)take((size_t)NNODES * 128 * 2);
  half_t* Z16  = (half_t*)take((size_t)NNODES * 64 * 2);
  half_t* BT1  = (half_t*)take((size_t)512 * 128 * 2);
  half_t* BT2  = (half_t*)take((size_t)512 * 128 * 2);
  half_t* BTC1 = (half_t*)take((size_t)128 * 128 * 2);
  half_t* BTC2 = (half_t*)take((size_t)64 * 128 * 2);
  half_t* BTC3 = (half_t*)take((size_t)48 * 64 * 2);
  float*  H    = (float*)take((size_t)NNODES * 512 * 4);
  float*  OUT1 = (float*)take((size_t)NNODES * 128 * 4);
  float*  OUT2 = (float*)take((size_t)NNODES * 128 * 4);
  unsigned* AGG = (unsigned*)take((size_t)3 * NNODES * 128 * 4);
  // classifier temporaries alias H (H is dead after conv2 aggregation reads it)
  float* C1 = H;                                   // [NNODES][128]
  float* C2 = (float*)((char*)H + (size_t)NNODES * 128 * 4 * 2);  // [NNODES][64]
  float* C3 = (float*)((char*)H + (size_t)NNODES * 128 * 4 * 3);  // [NNODES][48]

  const int TPB = 256;
  // --- prep: f16 activations + transposed f16 weights ---
  f32_to_f16_k<<<(NNODES * 128) / TPB, TPB, 0, stream>>>(x, XH, (size_t)NNODES * 128);
  pack_wT_k<<<128, 128, 0, stream>>>(w1_root, BT1, 128, 128);
  for (int r = 0; r < 3; r++)
    pack_wT_k<<<128, 128, 0, stream>>>(w1_rel + (size_t)r * 128 * 128,
                                       BT1 + (size_t)(128 + 128 * r) * 128, 128, 128);
  pack_wT_k<<<128, 128, 0, stream>>>(w2_root, BT2, 128, 128);
  for (int r = 0; r < 3; r++)
    pack_wT_k<<<128, 128, 0, stream>>>(w2_rel + (size_t)r * 128 * 128,
                                       BT2 + (size_t)(128 + 128 * r) * 128, 128, 128);
  pack_wT_k<<<128, 128, 0, stream>>>(cw1, BTC1, 128, 128);
  pack_wT_k<<<64, 128, 0, stream>>>(cw2, BTC2, 128, 64);
  pack_wT_k<<<48, 64, 0, stream>>>(cw3, BTC3, 64, 40);

  // --- conv1: H = x @ [root|rel0|rel1|rel2] ; add-aggregate ; LN+ReLU ---
  gemm_wmma_k<4><<<3125, TPB, 0, stream>>>(XH, BT1, H, NNODES, 512, 128);
  conv_init_k<<<(NNODES * 128) / TPB, TPB, 0, stream>>>(H, b1, OUT1, NNODES);
  scatter_add_k<<<NEDGES / 8, TPB, 0, stream>>>(H, ei, et, OUT1, NEDGES);
  ln_relu_k<<<NNODES / 8, TPB, 0, stream>>>(OUT1, X1H, ln1_g, ln1_b, NNODES);

  // --- conv2: H = x1 @ [root|rel] ; per-relation max-aggregate ; LN+ReLU + 0.2*x1 ---
  gemm_wmma_k<4><<<3125, TPB, 0, stream>>>(X1H, BT2, H, NNODES, 512, 128);
  conv_init_k<<<(NNODES * 128) / TPB, TPB, 0, stream>>>(H, b2, OUT2, NNODES);
  fill_u32_k<<<(3 * NNODES * 128) / TPB, TPB, 0, stream>>>(AGG, ENC_NEG_INF,
                                                           (size_t)3 * NNODES * 128);
  scatter_max_k<<<NEDGES / 8, TPB, 0, stream>>>(H, ei, et, AGG, NEDGES, NNODES);
  conv2_fin_k<<<NNODES / 8, TPB, 0, stream>>>(OUT2, AGG, OUT1, ln2_g, ln2_b, H16C, NNODES);

  // --- classifier MLP ---
  gemm_wmma_k<4><<<782, TPB, 0, stream>>>(H16C, BTC1, C1, NNODES, 128, 128); // 6250 tiles
  ln_gelu_k<128><<<NNODES / 8, TPB, 0, stream>>>(C1, cb1, cln1_g, cln1_b, Y16, NNODES);
  gemm_wmma_k<4><<<391, TPB, 0, stream>>>(Y16, BTC2, C2, NNODES, 64, 128);   // 3125 tiles
  ln_gelu_k<64><<<NNODES / 8, TPB, 0, stream>>>(C2, cb2, cln2_g, cln2_b, Z16, NNODES);
  gemm_wmma_k<3><<<391, TPB, 0, stream>>>(Z16, BTC3, C3, NNODES, 48, 64);    // 3125 tiles
  lsm_k<<<NNODES / 8, TPB, 0, stream>>>(C3, cb3, out, NNODES);
  (void)in_sizes; (void)n_in; (void)out_size; (void)ws_size;
}
